// LstmLayer_49246095016105
// MI455X (gfx1250) — matrix operand order repaired and checked by reading, not compiled, not run on hardware
//
#include <hip/hip_runtime.h>
#include <hip/hip_bf16.h>

// LSTM: data (64,2048,128) f32, W (256,512) f32, b (512) f32 -> out (64,2048,128) f32
// 4 self-contained blocks (16 batch rows), 16 waves/block, W f16 resident in VGPRs.
//   z = [x_t, h_{t-1}] @ W + b  via v_wmma_f32_16x16x32_f16, native v_tanh gates.

#define B_   64
#define T_   2048
#define D_   128
#define U_   128
#define G_   512
#define KTOT 256           // D_ + U_
#define ACT_STRIDE 264     // 256 + 8 halves pad (528B stride -> conflict-free b128)
#define P_STRIDE   132     // 128 + 4 f32 pad

typedef __attribute__((ext_vector_type(16))) _Float16 v16h;
typedef __attribute__((ext_vector_type(8)))  _Float16 v8h;
typedef __attribute__((ext_vector_type(8)))  float    v8f;

union FragU { v16h v; v8h h[2]; };
union HPack { v8h v; _Float16 e[8]; };

__global__ void wt_convert_kernel(const float* __restrict__ W,
                                  _Float16* __restrict__ WT) {
    int tid = blockIdx.x * blockDim.x + threadIdx.x;   // 0..131071
    int k = tid >> 9;          // 0..255
    int n = tid & 511;         // 0..511  (coalesced read of W[k][:])
    WT[n * KTOT + k] = (_Float16)W[k * G_ + n];
}

// Branch-free transcendentals (native v_tanh_f32 on CDNA5).
__device__ __forceinline__ float fast_tanh(float x) {
#if __has_builtin(__builtin_amdgcn_tanhf)
    return __builtin_amdgcn_tanhf(x);
#else
    return 1.0f - 2.0f * __builtin_amdgcn_rcpf(
               __builtin_amdgcn_exp2f(2.885390081777927f * x) + 1.0f);
#endif
}
// sigmoid(z + b) with half-bias hb = 0.5*b folded into the tanh argument fma.
__device__ __forceinline__ float sigm_b(float z, float hb) {
    return 0.5f + 0.5f * fast_tanh(__builtin_fmaf(0.5f, z, hb));
}

__global__ __launch_bounds__(512) void lstm_scan_kernel(
    const float* __restrict__ data,   // (64,2048,128)
    const float* __restrict__ bias,   // (512)
    const _Float16* __restrict__ WT,  // (512,256) f16, gate-column-major
    float* __restrict__ out)          // (64,2048,128)
{
    __shared__ __align__(16) _Float16 act[16 * ACT_STRIDE]; // [row][k] : x_t | h_{t-1}
    __shared__ __align__(16) float    Pb[16 * P_STRIDE];    // sig(i)*tanh(g) exchange

    const int tid   = threadIdx.x;
    const int wave  = tid >> 5;
    const int lane  = tid & 31;
    const int ln15  = lane & 15;
    const int hi    = lane >> 4;
    const int batch0 = blockIdx.x * 16;

    // waves 0..7 -> (i, g) tiles of unit-group ug; waves 8..15 -> (f, o).
    const int ug = wave & 7;
    int col0_t0, col0_t1;
    if (wave < 8) { col0_t0 = 16 * ug;        col0_t1 = 384 + 16 * ug; }  // i, g
    else          { col0_t0 = 128 + 16 * ug;  col0_t1 = 256 + 16 * ug; }  // f, o

    // ---- Persistent B fragments (time-invariant weights) in registers ----
    FragU bf[2][8];
    #pragma unroll
    for (int j = 0; j < 2; ++j) {
        const int ncol = (j ? col0_t1 : col0_t0) + ln15;
        const _Float16* base = WT + (size_t)ncol * KTOT;
        #pragma unroll
        for (int kb = 0; kb < 8; ++kb) {
            const v8h* g0 = (const v8h*)(base + kb * 32 + hi * 8);
            bf[j][kb].h[0] = g0[0];
            bf[j][kb].h[1] = g0[2];
        }
    }

    // Bias scalars folded into gate math (no broadcast C vectors needed).
    const float b0 = bias[col0_t0 + ln15];
    const float b1 = bias[col0_t1 + ln15];
    const float hb0 = 0.5f * b0;
    const float bg  = (wave < 8) ? b1 : 0.5f * b1;

    float c_[8];
    #pragma unroll
    for (int r = 0; r < 8; ++r) c_[r] = 1.0f;   // reference c0 = ones

    float hreg[8];                               // h_t staged for deferred store
    #pragma unroll
    for (int r = 0; r < 8; ++r) hreg[r] = 0.0f;

    // Per-thread staging coords (waves 0..7 / first 256 threads)
    const int srow = (tid & 255) >> 4;          // 0..15
    const int scol = (tid & 15) * 8;            // 0,8,..,120

    // ---- Init LDS: waves 0..7 load x_0; waves 8..15 zero h region ----
    if (wave < 8) {
        const float* src = data + ((size_t)(batch0 + srow) * T_) * D_ + scol;
        float4 xa = *(const float4*)(src);
        float4 xb = *(const float4*)(src + 4);
        HPack hp;
        hp.e[0]=(_Float16)xa.x; hp.e[1]=(_Float16)xa.y; hp.e[2]=(_Float16)xa.z; hp.e[3]=(_Float16)xa.w;
        hp.e[4]=(_Float16)xb.x; hp.e[5]=(_Float16)xb.y; hp.e[6]=(_Float16)xb.z; hp.e[7]=(_Float16)xb.w;
        *(v8h*)(act + srow * ACT_STRIDE + scol) = hp.v;
    } else {
        HPack hz;
        #pragma unroll
        for (int q = 0; q < 8; ++q) hz.e[q] = (_Float16)0.0f;
        *(v8h*)(act + srow * ACT_STRIDE + 128 + scol) = hz.v;
    }
    __syncthreads();

    const int colU = ug * 16 + ln15;           // unit column 0..127

    // ---- Sequential scan ----
    for (int t = 0; t < T_; ++t) {
        // Deferred output store: h_{t-1} goes out while this step's GEMM runs.
        if (wave >= 8 && t > 0) {
            #pragma unroll
            for (int r = 0; r < 8; ++r) {
                int m = r + 8 * hi;
                __builtin_nontemporal_store(
                    hreg[r], &out[((size_t)(batch0 + m) * T_ + (t - 1)) * U_ + colU]);
            }
        }

        v8f acc0 = {0.f,0.f,0.f,0.f,0.f,0.f,0.f,0.f};
        v8f acc1 = {0.f,0.f,0.f,0.f,0.f,0.f,0.f,0.f};

        // Preload 6 A-fragments (12 ds_load_b128 in flight), pinned above the
        // WMMA chain by a full scheduling barrier; fragments 6,7 stream into
        // recycled slots 0,1 (WAR pins each reload after its kb 0/1 consumer).
        const _Float16* arow = act + ln15 * ACT_STRIDE + hi * 8;
        FragU af[6];
        #pragma unroll
        for (int p = 0; p < 6; ++p) {
            const _Float16* ab = arow + p * 32;
            af[p].h[0] = *(const v8h*)(ab);
            af[p].h[1] = *(const v8h*)(ab + 16);
        }
#if __has_builtin(__builtin_amdgcn_sched_barrier)
        __builtin_amdgcn_sched_barrier(0);
#endif
        #pragma unroll
        for (int kb = 0; kb < 8; ++kb) {
            const int slot = (kb < 6) ? kb : (kb - 6);
            acc1 = __builtin_amdgcn_wmma_f32_16x16x32_f16(
                false, af[slot].v, false, bf[1][kb].v, (short)0, acc1, false, false);
            acc0 = __builtin_amdgcn_wmma_f32_16x16x32_f16(
                false, af[slot].v, false, bf[0][kb].v, (short)0, acc0, false, false);
            if (kb < 2) {
                const _Float16* ab = arow + (kb + 6) * 32;
                af[kb].h[0] = *(const v8h*)(ab);
                af[kb].h[1] = *(const v8h*)(ab + 16);
            }
        }

        float4 xa, xb;          // x_{t+1} staged through registers (waves 0..7)
        float sf[8], so_[8];    // pre-barrier sigmoids            (waves 8..15)
        if (wave < 8) {
            if (t + 1 < T_) {
                // Global loads issued here; latency overlaps P-exchange + h-update.
                const float* src = data + ((size_t)(batch0 + srow) * T_ + (t + 1)) * D_ + scol;
                xa = *(const float4*)(src);
                xb = *(const float4*)(src + 4);
                if (t + 4 < T_)
                    __builtin_prefetch(src + (size_t)3 * D_, 0, 0);  // t+4 into GL2
            }
            #pragma unroll
            for (int r = 0; r < 8; ++r) {
                int m = r + 8 * hi;
                // p = sigmoid(z_i + b_i) * tanh(z_g + b_g)
                Pb[m * P_STRIDE + colU] =
                    sigm_b(acc0[r], hb0) * fast_tanh(acc1[r] + bg);
            }
        } else {
            // Overlap with waves 0..7's P chain: sig(f), sig(o) need only our
            // own accumulators, so compute them BEFORE the barrier.
            #pragma unroll
            for (int r = 0; r < 8; ++r) {
                sf[r]  = sigm_b(acc0[r], hb0);   // sigmoid(z_f + b_f)
                so_[r] = sigm_b(acc1[r], bg);    // sigmoid(z_o + b_o)
            }
        }
        __syncthreads();   // A-frags consumed; P ready

        if (wave >= 8) {
            #pragma unroll
            for (int r = 0; r < 8; ++r) {
                int m = r + 8 * hi;
                float p  = Pb[m * P_STRIDE + colU];
                float cv = sf[r] * c_[r] + p;            // sig(f)*c + sig(i)*tanh(g)
                c_[r] = cv;
                float h = so_[r] * fast_tanh(cv);        // sig(o)*tanh(c)
                hreg[r] = h;                                   // deferred global store
                act[m * ACT_STRIDE + 128 + colU] = (_Float16)h; // feedback for t+1
            }
        } else if (t + 1 < T_) {
            HPack hp;
            hp.e[0]=(_Float16)xa.x; hp.e[1]=(_Float16)xa.y; hp.e[2]=(_Float16)xa.z; hp.e[3]=(_Float16)xa.w;
            hp.e[4]=(_Float16)xb.x; hp.e[5]=(_Float16)xb.y; hp.e[6]=(_Float16)xb.z; hp.e[7]=(_Float16)xb.w;
            *(v8h*)(act + srow * ACT_STRIDE + scol) = hp.v;
        }
        __syncthreads();   // h_t and x_{t+1} visible for next step
    }

    // Final deferred store (t = T-1)
    if (wave >= 8) {
        #pragma unroll
        for (int r = 0; r < 8; ++r) {
            int m = r + 8 * hi;
            __builtin_nontemporal_store(
                hreg[r], &out[((size_t)(batch0 + m) * T_ + (T_ - 1)) * U_ + colU]);
        }
    }
}

extern "C" void kernel_launch(void* const* d_in, const int* in_sizes, int n_in,
                              void* d_out, int out_size, void* d_ws, size_t ws_size,
                              hipStream_t stream) {
    (void)in_sizes; (void)n_in; (void)out_size; (void)ws_size;
    const float* data = (const float*)d_in[0];   // (64,2048,128) f32
    const float* W    = (const float*)d_in[1];   // (256,512) f32
    const float* b    = (const float*)d_in[2];   // (512) f32
    float* out        = (float*)d_out;           // (64,2048,128) f32
    _Float16* WT      = (_Float16*)d_ws;         // 512*256 f16 = 256 KB scratch

    wt_convert_kernel<<<512, 256, 0, stream>>>(W, WT);
    lstm_scan_kernel<<<4, 512, 0, stream>>>(data, b, WT, out);
}
